// TextEncoder_67405216743597
// MI455X (gfx1250) — matrix operand order repaired and checked by reading, not compiled, not run on hardware
//
#include <hip/hip_runtime.h>
#include <hip/hip_bf16.h>

// ---------------------------------------------------------------------------
// MI455X (gfx1250) TextEncoder: BoW + BiGRU (persistent WMMA scan) + Conv1d
// All matmuls via v_wmma_f32_16x16x32_bf16 (f32 accumulate).
// ---------------------------------------------------------------------------

typedef __attribute__((ext_vector_type(16))) __bf16 v16bf;
typedef __attribute__((ext_vector_type(8)))  __bf16 v8bf;
typedef __attribute__((ext_vector_type(8)))  float  v8f;

#define B_  32
#define L_  512
#define V_  32000
#define E_  500
#define H_  1024
#define G3  3072            // 3*H
#define ML  (B_ * L_)       // 16384 rows
#define OUTSTRIDE (V_ + 2*H_ + 1536)
#define SCAN_G 8            // workgroups per direction in the scan

// workspace byte offsets
#define OFF_GXF   0ull
#define OFF_GXB   (OFF_GXF  + (size_t)ML * G3 * 4)          // 201326592
#define OFF_HID   (OFF_GXB  + (size_t)ML * G3 * 4)
#define OFF_HF32  (OFF_HID  + (size_t)ML * 2 * H_ * 4)      // hiddens (B,L,2H) f32
#define OFF_HBF   (OFF_HF32 + (size_t)2 * 2 * B_ * H_ * 4)  // h double buffer f32
#define OFF_BAR   (OFF_HBF  + (size_t)2 * 2 * B_ * H_ * 2)  // h double buffer bf16
#define OFF_WHHB  (OFF_BAR  + 256)                          // whh in bf16, [2][3072][1024]

// ---------------- helpers ----------------

__global__ void k_zero(float* __restrict__ p, int n) {
  int stride = gridDim.x * blockDim.x;
  for (int i = blockIdx.x * blockDim.x + threadIdx.x; i < n; i += stride) p[i] = 0.0f;
}

__global__ void k_cvt_bf16(const float* __restrict__ src, __bf16* __restrict__ dst, int n) {
  int stride = gridDim.x * blockDim.x;
  for (int i = blockIdx.x * blockDim.x + threadIdx.x; i < n; i += stride)
    dst[i] = (__bf16)src[i];
}

__global__ void k_bow(const int* __restrict__ x, float* __restrict__ out) {
  int id = blockIdx.x * blockDim.x + threadIdx.x;
  if (id >= ML) return;
  int b = id >> 9;
  int v = x[id];
  if (v != 0) atomicAdd(&out[(size_t)b * OUTSTRIDE + v], 1.0f);  // PAD col stays 0
}

__global__ void k_rnn_sum(const float* __restrict__ hid, float* __restrict__ out) {
  int id = blockIdx.x * blockDim.x + threadIdx.x;   // 32*2048
  if (id >= B_ * 2 * H_) return;
  int b = id / (2 * H_), c = id % (2 * H_);
  float s = 0.0f;
  for (int t = 0; t < L_; ++t) s += hid[((size_t)(b * L_ + t)) * (2 * H_) + c];
  out[(size_t)b * OUTSTRIDE + V_ + c] = s * (1.0f / (float)L_);
}

// A fragment (16x32 bf16) from a row pointer into an [m][32] tile (k-contig)
__device__ __forceinline__ v16bf a_frag(const __bf16* arow, int lane) {
  const int kA = (lane & 16) ? 16 : 0;
  v8bf lo = *(const v8bf*)(arow + kA);
  v8bf hi = *(const v8bf*)(arow + kA + 8);
  return __builtin_shufflevector(lo, hi, 0,1,2,3,4,5,6,7,8,9,10,11,12,13,14,15);
}

// ---------------- input-gate GEMM: gx = emb[x] @ wih^T + bih (both dirs) -----
// grid (24, 128, 2), block 256; 128x128 tile per WG, K=512 (E padded).

__global__ __launch_bounds__(256) void k_gemm_gx(
    const int* __restrict__ x, const float* __restrict__ emb,
    const float* __restrict__ wih_f, const float* __restrict__ wih_b,
    const float* __restrict__ bih_f, const float* __restrict__ bih_b,
    float* __restrict__ gxf, float* __restrict__ gxb)
{
  __shared__ __bf16 At[128 * 32];   // [m][k]
  __shared__ __bf16 Bt[128 * 32];   // [n][k]
  const int dir = blockIdx.z;
  const float* wih = dir ? wih_b : wih_f;
  const float* bih = dir ? bih_b : bih_f;
  float* gx = dir ? gxb : gxf;
  const int m0 = blockIdx.y * 128;
  const int n0 = blockIdx.x * 128;
  const int tid = threadIdx.x, lane = tid & 31, wv = tid >> 5;

  v8f zero = {};
  v8f acc[8];
#pragma unroll
  for (int i = 0; i < 8; ++i) acc[i] = zero;

  const int lrow  = tid >> 1;
  const int khalf = (tid & 1) * 16;
  const int token = x[m0 + lrow];

  for (int kt = 0; kt < 16; ++kt) {
    const int k0 = kt * 32;
#pragma unroll
    for (int q = 0; q < 16; ++q) {
      int k = k0 + khalf + q;
      At[lrow * 32 + khalf + q] = (__bf16)((k < E_) ? emb[(size_t)token * E_ + k] : 0.0f);
    }
#pragma unroll
    for (int q = 0; q < 16; ++q) {
      int k = k0 + khalf + q;
      Bt[lrow * 32 + khalf + q] = (__bf16)((k < E_) ? wih[(size_t)(n0 + lrow) * E_ + k] : 0.0f);
    }
    if (kt < 15) __builtin_prefetch(&wih[(size_t)(n0 + lrow) * E_ + k0 + 32], 0, 1);
    __syncthreads();

    v16bf a = a_frag(&At[(wv * 16 + (lane & 15)) * 32], lane);
    const int kB = (lane & 16) ? 16 : 0;
#pragma unroll
    for (int nt = 0; nt < 8; ++nt) {
      v16bf b = *(const v16bf*)&Bt[(nt * 16 + (lane & 15)) * 32 + kB];
      acc[nt] = __builtin_amdgcn_wmma_f32_16x16x32_bf16(
          false, a, false, b, (short)0, acc[nt], false, false);
    }
    __syncthreads();
  }

  const int rbase = m0 + wv * 16 + ((lane & 16) ? 8 : 0);
#pragma unroll
  for (int nt = 0; nt < 8; ++nt) {
    const int col = n0 + nt * 16 + (lane & 15);
    const float bias = bih[col];
#pragma unroll
    for (int k = 0; k < 8; ++k)
      gx[(size_t)(rbase + k) * G3 + col] = acc[nt][k] + bias;
  }
}

// ---------------- persistent bidirectional GRU scan --------------------------
// grid 16 (= 2 dirs x 8 groups), block 256. Each group owns 128 h-columns.
// Per step: gh(32x384) = h(32x1024, bf16) @ whh_slice^T via WMMA; gate math in
// f32; software barrier across the 8 groups of a direction.

__global__ __launch_bounds__(256) void k_scan(
    const float* __restrict__ gxf, const float* __restrict__ gxb,
    const __bf16* __restrict__ whhb,          // [2][3072][1024] bf16
    const float* __restrict__ bhh_f, const float* __restrict__ bhh_b,
    const int* __restrict__ lengths,
    float* __restrict__ hf32,                 // [2][2][32][1024]
    __bf16* __restrict__ hbf,                 // [2][2][32][1024]
    float* __restrict__ hiddens,              // [32][512][2048]
    unsigned* __restrict__ bars)
{
  __shared__ float ghs[B_ * 384];
  __shared__ int s_len[B_];
  const int tid = threadIdx.x, lane = tid & 31, wv = tid >> 5;
  const int dir = blockIdx.x / SCAN_G;
  const int grp = blockIdx.x % SCAN_G;
  const int c0  = grp * 128;
  const float* gx   = dir ? gxb : gxf;
  const __bf16* whh = whhb + (size_t)dir * G3 * H_;
  const float* bhh  = dir ? bhh_b : bhh_f;
  unsigned* cnt = bars + dir * 32;
  unsigned* gen = bars + dir * 32 + 16;
  if (tid < B_) s_len[tid] = lengths[tid];
  __syncthreads();

  const int ml = lane & 15;
  const int kA = (lane & 16) ? 16 : 0;
  const int kB = (lane & 16) ? 16 : 0;
  v8f zero = {};

  for (int t = 0; t < L_; ++t) {
    const int p = t & 1;
    const float*  hprevf = hf32 + (size_t)(p * 2 + dir) * B_ * H_;
    float*        hnextf = hf32 + (size_t)((p ^ 1) * 2 + dir) * B_ * H_;
    const __bf16* hprevb = hbf  + (size_t)(p * 2 + dir) * B_ * H_;
    __bf16*       hnextb = hbf  + (size_t)((p ^ 1) * 2 + dir) * B_ * H_;

    v8f acc[2][3];
#pragma unroll
    for (int i = 0; i < 2; ++i)
#pragma unroll
      for (int j = 0; j < 3; ++j) acc[i][j] = zero;

    for (int kt = 0; kt < 32; ++kt) {
      const int k0 = kt * 32;
      v16bf a[2];
#pragma unroll
      for (int mt = 0; mt < 2; ++mt) {
        const __bf16* ar = hprevb + (size_t)(mt * 16 + ml) * H_ + k0;
        v8bf lo = *(const v8bf*)(ar + kA);
        v8bf hi = *(const v8bf*)(ar + kA + 8);
        a[mt] = __builtin_shufflevector(lo, hi, 0,1,2,3,4,5,6,7,8,9,10,11,12,13,14,15);
      }
#pragma unroll
      for (int j = 0; j < 3; ++j) {
        const int nt = wv * 3 + j;                                   // 0..23
        const int g  = (nt >> 3) * H_ + c0 + (nt & 7) * 16 + ml;     // gate row
        const __bf16* bp = whh + (size_t)g * H_ + k0 + kB;
        if (kt < 31) __builtin_prefetch(bp + 32, 0, 1);
        v16bf b = *(const v16bf*)bp;
#pragma unroll
        for (int mt = 0; mt < 2; ++mt)
          acc[mt][j] = __builtin_amdgcn_wmma_f32_16x16x32_bf16(
              false, a[mt], false, b, (short)0, acc[mt][j], false, false);
      }
    }

    // spill gh to LDS in (b, local-col) layout
#pragma unroll
    for (int j = 0; j < 3; ++j) {
      const int col = (wv * 3 + j) * 16 + ml;
#pragma unroll
      for (int mt = 0; mt < 2; ++mt) {
        const int rb = mt * 16 + ((lane & 16) ? 8 : 0);
#pragma unroll
        for (int k = 0; k < 8; ++k)
          ghs[(rb + k) * 384 + col] = acc[mt][j][k];
      }
    }
    __syncthreads();

    // elementwise gate math: 32x128 elements / 256 threads
#pragma unroll
    for (int i = 0; i < 16; ++i) {
      const int e = tid + 256 * i;
      const int b = e >> 7;
      const int j = e & 127;
      const int c = c0 + j;
      const int len = s_len[b];
      int row = dir ? (len - 1 - t) : t;
      if (row < 0) row = 0;
      const float* gxr = gx + (size_t)(b * L_ + row) * G3;
      const float ghr = ghs[b * 384 + j]       + bhh[c];
      const float ghz = ghs[b * 384 + 128 + j] + bhh[H_ + c];
      const float ghn = ghs[b * 384 + 256 + j] + bhh[2 * H_ + c];
      const float r = 1.0f / (1.0f + expf(-(gxr[c] + ghr)));
      const float z = 1.0f / (1.0f + expf(-(gxr[H_ + c] + ghz)));
      const float n = tanhf(gxr[2 * H_ + c] + r * ghn);
      const float hp = hprevf[b * H_ + c];
      const float hn = (1.0f - z) * n + z * hp;
      hnextf[b * H_ + c] = hn;
      hnextb[b * H_ + c] = (__bf16)hn;
      if (t < len) {
        const int tout = dir ? (len - 1 - t) : t;
        hiddens[(size_t)(b * L_ + tout) * (2 * H_) + dir * H_ + c] = hn;
      }
    }

    // software barrier across this direction's 8 groups
    __threadfence();
    __syncthreads();
    if (tid == 0) {
      const unsigned target = (unsigned)(t + 1);
      if (atomicAdd(cnt, 1u) == (unsigned)(SCAN_G - 1)) {
        atomicExch(cnt, 0u);
        __threadfence();
        atomicExch(gen, target);
      } else {
        while (atomicAdd(gen, 0u) < target) __builtin_amdgcn_s_sleep(2);
        __threadfence();
      }
    }
    __syncthreads();
  }
}

// ---------------- Conv1d branch as shifted WMMA GEMMs + ReLU + max ----------
// grid (4, 128, 3) -> z selects k=2/3/4; 128x128 tile; K = 2048 * ks.

__global__ __launch_bounds__(256) void k_conv(
    const float* __restrict__ hid,
    const float* __restrict__ w0, const float* __restrict__ w1, const float* __restrict__ w2,
    const float* __restrict__ cb0, const float* __restrict__ cb1, const float* __restrict__ cb2,
    float* __restrict__ out)
{
  __shared__ __bf16 At[128 * 32];
  __shared__ __bf16 Bt[128 * 32];
  const int z = blockIdx.z;
  const float* w  = (z == 0) ? w0  : (z == 1) ? w1  : w2;
  const float* cb = (z == 0) ? cb0 : (z == 1) ? cb1 : cb2;
  const int ks = z + 2;
  const int pl = (ks - 1) >> 1;        // torch 'same' left pad: k=2 -> 0, k=3,4 -> 1
  const int obase = z * 512;
  const int m0 = blockIdx.y * 128;     // 128 rows, all in one batch (128 | 512)
  const int n0 = blockIdx.x * 128;
  const int b  = m0 >> 9;
  const int tb = m0 & 511;
  const int tid = threadIdx.x, lane = tid & 31, wv = tid >> 5;

  v8f zero = {};
  v8f acc[8];
#pragma unroll
  for (int i = 0; i < 8; ++i) acc[i] = zero;

  const int lrow  = tid >> 1;
  const int khalf = (tid & 1) * 16;

  for (int kk = 0; kk < ks; ++kk) {
    const int tsrc = tb + lrow + kk - pl;
    const bool inb = (tsrc >= 0) && (tsrc < L_);
    const float* hrow = hid + (size_t)(b * L_ + (inb ? tsrc : 0)) * (2 * H_);
    for (int ct = 0; ct < 64; ++ct) {
      const int c0 = ct * 32;
#pragma unroll
      for (int q = 0; q < 16; ++q)
        At[lrow * 32 + khalf + q] = (__bf16)(inb ? hrow[c0 + khalf + q] : 0.0f);
#pragma unroll
      for (int q = 0; q < 16; ++q)
        Bt[lrow * 32 + khalf + q] =
            (__bf16)w[(size_t)(n0 + lrow) * (2 * H_) * ks + (size_t)(c0 + khalf + q) * ks + kk];
      __syncthreads();

      v16bf a = a_frag(&At[(wv * 16 + (lane & 15)) * 32], lane);
      const int kB = (lane & 16) ? 16 : 0;
#pragma unroll
      for (int nt = 0; nt < 8; ++nt) {
        v16bf bfr = *(const v16bf*)&Bt[(nt * 16 + (lane & 15)) * 32 + kB];
        acc[nt] = __builtin_amdgcn_wmma_f32_16x16x32_bf16(
            false, a, false, bfr, (short)0, acc[nt], false, false);
      }
      __syncthreads();
    }
  }

  // bias + ReLU + max over this wave's 16 rows, then atomic max into output
#pragma unroll
  for (int nt = 0; nt < 8; ++nt) {
    const int o = n0 + nt * 16 + (lane & 15);
    const float bias = cb[o];
    float m = 0.0f;                                   // ReLU floor
#pragma unroll
    for (int k = 0; k < 8; ++k) m = fmaxf(m, acc[nt][k] + bias);
    m = fmaxf(m, __shfl_xor(m, 16, 32));              // combine M halves
    if (lane < 16) {
      float* dst = out + (size_t)b * OUTSTRIDE + V_ + 2 * H_ + obase + o;
      atomicMax((int*)dst, __float_as_int(m));        // valid: values >= 0
    }
  }
}

// ---------------- host side -------------------------------------------------

extern "C" void kernel_launch(void* const* d_in, const int* in_sizes, int n_in,
                              void* d_out, int out_size, void* d_ws, size_t ws_size,
                              hipStream_t stream) {
  const int*   x       = (const int*)d_in[0];
  const int*   lengths = (const int*)d_in[1];
  const float* emb     = (const float*)d_in[2];
  const float* wih_f   = (const float*)d_in[3];
  const float* whh_f   = (const float*)d_in[4];
  const float* bih_f   = (const float*)d_in[5];
  const float* bhh_f   = (const float*)d_in[6];
  const float* wih_b   = (const float*)d_in[7];
  const float* whh_b   = (const float*)d_in[8];
  const float* bih_b   = (const float*)d_in[9];
  const float* bhh_b   = (const float*)d_in[10];
  const float* cw0     = (const float*)d_in[11];
  const float* cb0     = (const float*)d_in[12];
  const float* cw1     = (const float*)d_in[13];
  const float* cb1     = (const float*)d_in[14];
  const float* cw2     = (const float*)d_in[15];
  const float* cb2     = (const float*)d_in[16];
  float* out = (float*)d_out;

  char* ws = (char*)d_ws;
  float*   gxf   = (float*)(ws + OFF_GXF);
  float*   gxb   = (float*)(ws + OFF_GXB);
  float*   hid   = (float*)(ws + OFF_HID);
  float*   hf32  = (float*)(ws + OFF_HF32);
  __bf16*  hbf   = (__bf16*)(ws + OFF_HBF);
  unsigned* bars = (unsigned*)(ws + OFF_BAR);
  __bf16*  whhb  = (__bf16*)(ws + OFF_WHHB);

  // 1) zero output, hiddens, h double-buffers (f32 + bf16) + barrier words
  k_zero<<<1024, 256, 0, stream>>>(out, B_ * OUTSTRIDE);
  k_zero<<<4096, 256, 0, stream>>>(hid, ML * 2 * H_);
  k_zero<<<256, 256, 0, stream>>>(hf32, (int)((OFF_WHHB - OFF_HF32) / 4));

  // 2) whh -> bf16 (one-time per call; L2-resident for the scan)
  k_cvt_bf16<<<2048, 256, 0, stream>>>(whh_f, whhb, G3 * H_);
  k_cvt_bf16<<<2048, 256, 0, stream>>>(whh_b, whhb + (size_t)G3 * H_, G3 * H_);

  // 3) bag of words
  k_bow<<<(ML + 255) / 256, 256, 0, stream>>>(x, out);

  // 4) input-gate GEMM for both directions (natural order; backward handled
  //    via per-step row indexing inside the scan)
  k_gemm_gx<<<dim3(24, 128, 2), 256, 0, stream>>>(x, emb, wih_f, wih_b,
                                                  bih_f, bih_b, gxf, gxb);

  // 5) persistent bidirectional GRU scan
  k_scan<<<2 * SCAN_G, 256, 0, stream>>>(gxf, gxb, whhb, bhh_f, bhh_b,
                                         lengths, hf32, hbf, hid, bars);

  // 6) rnn_emb = mean over padded length
  k_rnn_sum<<<256, 256, 0, stream>>>(hid, out);

  // 7) CNN branch: conv-as-GEMM + ReLU + global max
  k_conv<<<dim3(4, 128, 3), 256, 0, stream>>>(hid, cw0, cw1, cw2,
                                              cb0, cb1, cb2, out);

  (void)in_sizes; (void)n_in; (void)out_size; (void)ws_size;
}
